// PythonCodeGPT_10161892622594
// MI455X (gfx1250) — matrix operand order repaired and checked by reading, not compiled, not run on hardware
//
#include <hip/hip_runtime.h>

typedef __attribute__((ext_vector_type(16))) _Float16 v16h;
typedef __attribute__((ext_vector_type(8)))  float    v8f;
typedef unsigned int u32x4 __attribute__((ext_vector_type(4)));
typedef int          i32x4 __attribute__((ext_vector_type(4)));
typedef int          i32x8 __attribute__((ext_vector_type(8)));

#define D_MODEL 1024
#define NHEAD   16
#define HDIM    64
#define NLAYER  8
#define BATCH   4
#define SEQ     1024
#define VOCAB   32000
#define ROWS    (BATCH * SEQ)      // 4096
#define EPS     1e-5f

// ---------------------------------------------------------------------------
// Embedding: x[b,t,:] = tok_emb[idx[b,t],:] + pos_emb[t,:]   (f32 residual)
// ---------------------------------------------------------------------------
__global__ void embed_kernel(const int* __restrict__ idx,
                             const float* __restrict__ tok,
                             const float* __restrict__ pos,
                             float* __restrict__ x) {
  int row = blockIdx.x;
  int t = row & (SEQ - 1);
  int token = idx[row];
  const float* te = tok + (size_t)token * D_MODEL;
  const float* pe = pos + (size_t)t * D_MODEL;
  float* xo = x + (size_t)row * D_MODEL;
  for (int c = threadIdx.x; c < D_MODEL; c += blockDim.x)
    xo[c] = te[c] + pe[c];
}

// ---------------------------------------------------------------------------
// LayerNorm (f32 in, f16 out -> feeds WMMA GEMMs), one block per row
// ---------------------------------------------------------------------------
__global__ void ln_kernel(const float* __restrict__ x,
                          const float* __restrict__ g,
                          const float* __restrict__ b,
                          _Float16* __restrict__ o) {
  __shared__ float red[256];
  int row = blockIdx.x;
  const float* xr = x + (size_t)row * D_MODEL;
  float s = 0.f;
  for (int c = threadIdx.x; c < D_MODEL; c += 256) s += xr[c];
  red[threadIdx.x] = s; __syncthreads();
  for (int off = 128; off > 0; off >>= 1) {
    if (threadIdx.x < off) red[threadIdx.x] += red[threadIdx.x + off];
    __syncthreads();
  }
  float mean = red[0] * (1.f / D_MODEL);
  __syncthreads();
  float vs = 0.f;
  for (int c = threadIdx.x; c < D_MODEL; c += 256) {
    float d = xr[c] - mean; vs += d * d;
  }
  red[threadIdx.x] = vs; __syncthreads();
  for (int off = 128; off > 0; off >>= 1) {
    if (threadIdx.x < off) red[threadIdx.x] += red[threadIdx.x + off];
    __syncthreads();
  }
  float rstd = rsqrtf(red[0] * (1.f / D_MODEL) + EPS);
  _Float16* orow = o + (size_t)row * D_MODEL;
  for (int c = threadIdx.x; c < D_MODEL; c += 256)
    orow[c] = (_Float16)((xr[c] - mean) * rstd * g[c] + b[c]);
}

// ---------------------------------------------------------------------------
// Weight packing: f32 weight -> fragment-major packed f16.
// Logical element b(k,n) = W[(n/64)*b_blk + k*ldb + (n%64)].
// Packed layout: [kblk][ksub][ntile16][lane32][half16]; a GEMM B-tile is then
// two contiguous segments -> moved by the Tensor Data Mover.
// ---------------------------------------------------------------------------
__global__ void pack_b_kernel(const float* __restrict__ W, _Float16* __restrict__ P,
                              int Kd, int N, int ldb, long long b_blk,
                              long long in_ls, long long out_ls) {
  const float* Wz = W + (size_t)blockIdx.z * in_ls;
  _Float16* Pz = P + (size_t)blockIdx.z * out_ls;
  size_t total = (size_t)Kd * N;
  for (size_t idx = (size_t)blockIdx.x * blockDim.x + threadIdx.x; idx < total;
       idx += (size_t)gridDim.x * blockDim.x) {
    int k = (int)(idx / N), n = (int)(idx % N);
    float v = Wz[(size_t)(n >> 6) * b_blk + (size_t)k * ldb + (n & 63)];
    int ksub = (k & 63) >> 5, kk = k & 31;
    int lane = (n & 15) + ((kk >> 4) << 4);
    size_t off = (((((size_t)(k >> 6) * 2 + ksub) * (N >> 4) + (n >> 4)) << 5) + lane) * 16 +
                 (kk & 15);
    Pz[off] = (_Float16)v;
  }
}

// ---------------------------------------------------------------------------
// TDM: DMA two contiguous packed-B segments (8B elements, 2 rows) into LDS.
// D# per CDNA5 ISA: group0 = {count|flags, lds_addr,
// global_addr lo, global_addr hi | type=2}; group1 packs data_size=8B,
// huge tensor dims (no OOB), tile_dim0 = seg elems, tile_dim1 = 2, dim0 stride.
// Tracked by TENSORcnt; EXEC-independent, issued by wave 0 only (branch).
// ---------------------------------------------------------------------------
__device__ __forceinline__ void tdm_load_b_pair(unsigned lds_addr,
                                                unsigned long long gaddr,
                                                int tile0_elems,
                                                int row_stride_elems) {
  u32x4 g0 = {1u, lds_addr,
              (unsigned)(gaddr & 0xffffffffull),
              (unsigned)((gaddr >> 32) & 0x01ffffffull) | (2u << 30)};
  i32x8 g1 = {0x30000,                                  // data_size = 8B
              (int)0xFFFF0000u,                         // tensor_dim0 lo16
              (int)0xFFFF7FFFu,                         // tensor_dim0 hi / dim1 lo
              (int)(((unsigned)tile0_elems << 16) | 0x7FFFu),  // tile_dim0 | dim1 hi
              2,                                        // tile_dim1 = 2 rows
              row_stride_elems, 0, 0};                  // tensor_dim0_stride
  i32x4 gz = {0, 0, 0, 0};
#if __clang_major__ >= 23
  i32x8 gz8 = {0, 0, 0, 0, 0, 0, 0, 0};
  __builtin_amdgcn_tensor_load_to_lds(g0, g1, gz, gz, gz8, 0);
#else
  __builtin_amdgcn_tensor_load_to_lds(g0, g1, gz, gz, 0);
#endif
}

// ---------------------------------------------------------------------------
// WMMA GEMM, f16 operands, fragment-major LDS, TDM/direct staging, ping-pong.
//   BM=128 x BN(128|64), BK=64, 256 threads = 8 waves (4 rows x 2 cols).
//   BMODE: 0 = packed weights (TDM async DMA during compute)
//          1 = transposed activation, b(k,n)=B[n*ldb+k] (direct chunk copy)
//          2 = row-major activation (direct scalar staging; BN==64, PV GEMM)
// ---------------------------------------------------------------------------
__device__ __forceinline__ v8f wmma_f16(v16h a, v16h b, v8f c) {
  return __builtin_amdgcn_wmma_f32_16x16x32_f16(false, a, false, b, (short)0, c,
                                                false, false);
}

__device__ __forceinline__ void stage_A_direct(const _Float16* A, size_t base,
                                               int lda, _Float16* Ad, int tid) {
#pragma unroll
  for (int i = 0; i < 4; ++i) {
    int c = tid + i * 256;                 // 1024 chunks of 8 halves
    int row = c >> 3, w = c & 7;
    uint4 v = *(const uint4*)(A + base + (size_t)row * lda + w * 8);
    // k-run w*8..w*8+7: ksub=w>>2, hi=(w>>1)&1, halfbase=(w&1)*8
    int off = (((w >> 2) * 8 + (row >> 4)) * 32 + ((row & 15) + (((w >> 1) & 1) << 4))) * 16 +
              ((w & 1) << 3);
    *(uint4*)(Ad + off) = v;
  }
}

template <int BN, int BMODE>
__device__ __forceinline__ void stage_B_direct(const _Float16* B, int n0, int k0,
                                               int ldb, long long b_blk,
                                               _Float16* Bd, int tid) {
  if constexpr (BMODE == 1) {
#pragma unroll
    for (int i = 0; i < BN / 32; ++i) {
      int c = tid + i * 256;                 // BN*8 chunks
      int n = c >> 3, w = c & 7;
      uint4 v = *(const uint4*)(B + (size_t)(n0 + n) * ldb + k0 + w * 8);
      int off = (((w >> 2) * (BN / 16) + (n >> 4)) * 32 +
                 ((n & 15) + (((w >> 1) & 1) << 4))) * 16 + ((w & 1) << 3);
      *(uint4*)(Bd + off) = v;
    }
  } else if constexpr (BMODE == 2) {         // BN==64 only
#pragma unroll
    for (int i = 0; i < 16; ++i) {
      int e = tid + i * 256;                 // 64*BN elems
      int n = e & (BN - 1), k = e / BN;
      int gn = n0 + n;
      _Float16 v = B[(size_t)(gn >> 6) * b_blk + (size_t)(k0 + k) * ldb + (gn & 63)];
      int ksub = k >> 5, kk = k & 31;
      Bd[((ksub * (BN / 16) + (n >> 4)) * 32 + ((n & 15) + ((kk >> 4) << 4))) * 16 +
         (kk & 15)] = v;
    }
  }
}

template <int BN, int BMODE>
__global__ __launch_bounds__(256, 1)
void gemm_wmma(const _Float16* __restrict__ A, const _Float16* __restrict__ B,
               const float* __restrict__ bias, const float* __restrict__ R,
               void* __restrict__ Cv,
               int K, int N, int lda, int ldb, int ldc,
               long long a_bs, long long a_hs,
               long long b_bs, long long b_hs,
               long long c_bs, long long c_hs, int nh,
               long long b_blk, int out_f16,
               int relu, int causal, float scale) {
  constexpr int NJ = BN / 32;
  __shared__ v16h Asf[2][2 * 8 * 32];            // ping-pong [buf][ksub][mtile][lane]
  __shared__ v16h Bsf[2][2 * (BN / 16) * 32];

  int z = blockIdx.z;
  A += (size_t)(z / nh) * a_bs + (size_t)(z % nh) * a_hs;
  B += (size_t)(z / nh) * b_bs + (size_t)(z % nh) * b_hs;
  size_t coff = (size_t)(z / nh) * c_bs + (size_t)(z % nh) * c_hs;
  float* Cf = (float*)Cv + coff;
  _Float16* Ch = (_Float16*)Cv + coff;

  int n0 = blockIdx.x * BN, m0 = blockIdx.y * 128;
  int tid = threadIdx.x, lane = tid & 31, wave = tid >> 5;
  int wm = (wave & 3) * 32, wn = (wave >> 2) * (BN / 2);
  int hi_l = (lane >> 4) & 1, ln16 = lane & 15;

  // fully-masked causal tile: just write the mask value
  if (causal && n0 > m0 + 127) {
#pragma unroll
    for (int i = 0; i < 2; ++i)
#pragma unroll
      for (int j = 0; j < NJ; ++j)
#pragma unroll
        for (int r = 0; r < 8; ++r) {
          int gr = m0 + wm + 16 * i + r + 8 * hi_l;
          int gc = n0 + wn + 16 * j + ln16;
          Cf[(size_t)gr * ldc + gc] = -3.0e38f;
        }
    return;
  }

  v8f acc[2][NJ] = {};
  int wm16 = wm >> 4, wn16 = wn >> 4;
  int nkb = K >> 6;
  size_t abase = (size_t)m0 * lda;
  const int tdm_tile0 = BN * 8;                 // segment length in 8B elements
  const int tdm_stride = (N >> 4) * 128;        // ksub segment distance, 8B elems

  // prologue: stage k-block 0 into buffer 0
  stage_A_direct(A, abase, lda, (_Float16*)Asf[0], tid);
  if constexpr (BMODE == 0) {
    if (wave == 0) {
      unsigned long long ga = (unsigned long long)(uintptr_t)(B +
          (((size_t)0 * (N >> 4) + (n0 >> 4)) << 9));
      tdm_load_b_pair((unsigned)(uintptr_t)(void*)Bsf[0], ga, tdm_tile0, tdm_stride);
      __builtin_amdgcn_s_wait_tensorcnt(0);
    }
  } else {
    stage_B_direct<BN, BMODE>(B, n0, 0, ldb, b_blk, (_Float16*)Bsf[0], tid);
  }
  __syncthreads();

  for (int kb = 0; kb < nkb; ++kb) {
    int cur = kb & 1;
    bool more = (kb + 1 < nkb);
    if (more) {
      // warm next A tile in cache while computing (gfx1250 global_prefetch_b8)
      __builtin_prefetch(A + abase + (size_t)(tid >> 1) * lda + (size_t)(kb + 1) * 64, 0, 3);
      if constexpr (BMODE == 0) {
        if (wave == 0) {  // async DMA of next packed-B tile runs during compute
          unsigned long long ga = (unsigned long long)(uintptr_t)(B +
              ((((size_t)(kb + 1) * 2) * (N >> 4) + (n0 >> 4)) << 9));
          tdm_load_b_pair((unsigned)(uintptr_t)(void*)Bsf[cur ^ 1], ga,
                          tdm_tile0, tdm_stride);
        }
      } else if constexpr (BMODE == 1) {
        __builtin_prefetch(B + (size_t)(n0 + (tid >> 1)) * ldb + (size_t)(kb + 1) * 64, 0, 3);
      } else {
        __builtin_prefetch(B + (size_t)((kb + 1) * 64 + (tid & 63)) * ldb + (n0 & 63), 0, 3);
      }
    }

    const v16h* Af = Asf[cur];
    const v16h* Bf = Bsf[cur];
#pragma unroll
    for (int ks = 0; ks < 2; ++ks) {
      v16h a0 = Af[((ks << 3) + wm16 + 0) * 32 + lane];
      v16h a1 = Af[((ks << 3) + wm16 + 1) * 32 + lane];
      v16h bf[NJ];
#pragma unroll
      for (int j = 0; j < NJ; ++j)
        bf[j] = Bf[(ks * (BN / 16) + wn16 + j) * 32 + lane];
#pragma unroll
      for (int j = 0; j < NJ; ++j) {
        acc[0][j] = wmma_f16(a0, bf[j], acc[0][j]);
        acc[1][j] = wmma_f16(a1, bf[j], acc[1][j]);
      }
    }

    if (more) {  // direct-stage next A (short live ranges; cache-warm by now)
      stage_A_direct(A, abase + (size_t)(kb + 1) * 64, lda, (_Float16*)Asf[cur ^ 1], tid);
      if constexpr (BMODE != 0)
        stage_B_direct<BN, BMODE>(B, n0, (kb + 1) * 64, ldb, b_blk,
                                  (_Float16*)Bsf[cur ^ 1], tid);
      if constexpr (BMODE == 0)
        if (wave == 0) __builtin_amdgcn_s_wait_tensorcnt(0);
    }
    __syncthreads();
  }

  // epilogue; C/D layout: VGPR r -> row r (+8 for lanes 16-31), col = lane&15
#pragma unroll
  for (int i = 0; i < 2; ++i)
#pragma unroll
    for (int j = 0; j < NJ; ++j)
#pragma unroll
      for (int r = 0; r < 8; ++r) {
        int gr = m0 + wm + 16 * i + r + 8 * hi_l;
        int gc = n0 + wn + 16 * j + ln16;
        float v = acc[i][j][r];
        if (causal) {
          v = (gc > gr) ? -3.0e38f : v * scale;
          Cf[(size_t)gr * ldc + gc] = v;
        } else {
          if (bias) v += bias[gc];
          if (R)    v += R[(size_t)gr * ldc + gc];
          if (relu) v = fmaxf(v, 0.f);
          if (out_f16) Ch[(size_t)gr * ldc + gc] = (_Float16)v;
          else         Cf[(size_t)gr * ldc + gc] = v;
        }
      }
}

// ---------------------------------------------------------------------------
// Row softmax: f32 scores [B*H*T, T] -> f16 probs (PV GEMM A operand)
// ---------------------------------------------------------------------------
__global__ void softmax_kernel(const float* __restrict__ S, _Float16* __restrict__ P) {
  __shared__ float red[256];
  const float* sr = S + (size_t)blockIdx.x * SEQ;
  _Float16* pr = P + (size_t)blockIdx.x * SEQ;
  float m = -3.4e38f;
  for (int c = threadIdx.x; c < SEQ; c += 256) m = fmaxf(m, sr[c]);
  red[threadIdx.x] = m; __syncthreads();
  for (int off = 128; off > 0; off >>= 1) {
    if (threadIdx.x < off) red[threadIdx.x] = fmaxf(red[threadIdx.x], red[threadIdx.x + off]);
    __syncthreads();
  }
  m = red[0]; __syncthreads();
  float s = 0.f;
  for (int c = threadIdx.x; c < SEQ; c += 256) s += __expf(sr[c] - m);
  red[threadIdx.x] = s; __syncthreads();
  for (int off = 128; off > 0; off >>= 1) {
    if (threadIdx.x < off) red[threadIdx.x] += red[threadIdx.x + off];
    __syncthreads();
  }
  float inv = 1.f / red[0];
  for (int c = threadIdx.x; c < SEQ; c += 256)
    pr[c] = (_Float16)(__expf(sr[c] - m) * inv);
}

// ---------------------------------------------------------------------------
// Cross-entropy: loss += (logsumexp(row) - logit[target]) / (B*T)
// ---------------------------------------------------------------------------
__global__ void zero_loss_kernel(float* p) { *p = 0.f; }

__global__ void loss_kernel(const float* __restrict__ logits,
                            const int* __restrict__ tgt,
                            float* __restrict__ loss) {
  __shared__ float red[256];
  int row = blockIdx.x;
  const float* lr = logits + (size_t)row * VOCAB;
  float m = -3.4e38f;
  for (int c = threadIdx.x; c < VOCAB; c += 256) m = fmaxf(m, lr[c]);
  red[threadIdx.x] = m; __syncthreads();
  for (int off = 128; off > 0; off >>= 1) {
    if (threadIdx.x < off) red[threadIdx.x] = fmaxf(red[threadIdx.x], red[threadIdx.x + off]);
    __syncthreads();
  }
  m = red[0]; __syncthreads();
  float s = 0.f;
  for (int c = threadIdx.x; c < VOCAB; c += 256) s += __expf(lr[c] - m);
  red[threadIdx.x] = s; __syncthreads();
  for (int off = 128; off > 0; off >>= 1) {
    if (threadIdx.x < off) red[threadIdx.x] += red[threadIdx.x + off];
    __syncthreads();
  }
  if (threadIdx.x == 0) {
    float lse = m + __logf(red[0]);
    atomicAdd(loss, (lse - lr[tgt[row]]) * (1.f / (float)ROWS));
  }
}

// ---------------------------------------------------------------------------
// Host-side launch helper (three instantiations: 128/TDM, 128/trans, 64/scalar)
// ---------------------------------------------------------------------------
static inline void launch_gemm(hipStream_t s, const _Float16* A, const _Float16* B,
                               const float* bias, const float* R, void* C,
                               int M, int N, int K, int lda, int ldb, int ldc,
                               long long a_bs, long long a_hs,
                               long long b_bs, long long b_hs,
                               long long c_bs, long long c_hs,
                               int nh, int nz, int bn, long long b_blk,
                               int b_mode, int out_f16, int relu, int causal,
                               float scale) {
  dim3 grid(N / bn, M / 128, nz);
  if (bn == 128 && b_mode == 0)
    gemm_wmma<128, 0><<<grid, dim3(256), 0, s>>>(A, B, bias, R, C, K, N, lda, ldb, ldc,
                                                 a_bs, a_hs, b_bs, b_hs, c_bs, c_hs, nh,
                                                 b_blk, out_f16, relu, causal, scale);
  else if (bn == 128 && b_mode == 1)
    gemm_wmma<128, 1><<<grid, dim3(256), 0, s>>>(A, B, bias, R, C, K, N, lda, ldb, ldc,
                                                 a_bs, a_hs, b_bs, b_hs, c_bs, c_hs, nh,
                                                 b_blk, out_f16, relu, causal, scale);
  else
    gemm_wmma<64, 2><<<grid, dim3(256), 0, s>>>(A, B, bias, R, C, K, N, lda, ldb, ldc,
                                                a_bs, a_hs, b_bs, b_hs, c_bs, c_hs, nh,
                                                b_blk, out_f16, relu, causal, scale);
}

extern "C" void kernel_launch(void* const* d_in, const int* in_sizes, int n_in,
                              void* d_out, int out_size, void* d_ws, size_t ws_size,
                              hipStream_t stream) {
  (void)in_sizes; (void)n_in; (void)out_size; (void)ws_size;
  const int*   idx     = (const int*)  d_in[0];
  const int*   targets = (const int*)  d_in[1];
  const float* tok_emb = (const float*)d_in[2];
  const float* pos_emb = (const float*)d_in[3];
  const float* ln1_g   = (const float*)d_in[4];
  const float* ln1_b   = (const float*)d_in[5];
  const float* Wq      = (const float*)d_in[6];
  const float* Wk      = (const float*)d_in[7];
  const float* Wv      = (const float*)d_in[8];
  const float* Wo      = (const float*)d_in[9];
  const float* bo      = (const float*)d_in[10];
  const float* ln2_g   = (const float*)d_in[11];
  const float* ln2_b   = (const float*)d_in[12];
  const float* W1      = (const float*)d_in[13];
  const float* b1      = (const float*)d_in[14];
  const float* W2      = (const float*)d_in[15];
  const float* b2      = (const float*)d_in[16];
  const float* lnf_g   = (const float*)d_in[17];
  const float* lnf_b   = (const float*)d_in[18];
  const float* Wlm     = (const float*)d_in[19];
  const float* blm     = (const float*)d_in[20];

  const size_t XSZ  = (size_t)ROWS * D_MODEL;                 // 4096*1024
  const size_t SCSZ = (size_t)BATCH * NHEAD * SEQ * SEQ;      // [B,H,T,T]
  const long long WQ_LS = (long long)NHEAD * D_MODEL * HDIM;  // == D*D per layer
  const long long HBLK  = (long long)D_MODEL * HDIM;
  const float scale = 0.03125f;                               // D^-0.5 = 1/32

  // ---- workspace layout ----
  float* x      = (float*)d_ws;          // f32 residual stream
  float* scores = x + XSZ;               // f32 attention scores
  _Float16* hp  = (_Float16*)(scores + SCSZ);
  _Float16* hbuf = hp;  hp += XSZ;       // LN outputs (f16)
  _Float16* q    = hp;  hp += XSZ;
  _Float16* kbuf = hp;  hp += XSZ;
  _Float16* vbuf = hp;  hp += XSZ;
  _Float16* att  = hp;  hp += XSZ;
  _Float16* ffn  = hp;  hp += 4 * XSZ;
  _Float16* probs = hp; hp += SCSZ;      // f16 softmax probs
  _Float16* pWq  = hp;  hp += (size_t)NLAYER * D_MODEL * D_MODEL;
  _Float16* pWk  = hp;  hp += (size_t)NLAYER * D_MODEL * D_MODEL;
  _Float16* pWv  = hp;  hp += (size_t)NLAYER * D_MODEL * D_MODEL;
  _Float16* pWo  = hp;  hp += (size_t)NLAYER * D_MODEL * D_MODEL;
  _Float16* pW1  = hp;  hp += (size_t)NLAYER * D_MODEL * 4 * D_MODEL;
  _Float16* pW2  = hp;  hp += (size_t)NLAYER * 4 * D_MODEL * D_MODEL;
  _Float16* pWlm = hp;  hp += (size_t)D_MODEL * VOCAB;

  float* logits = (float*)d_out;
  float* lossp  = logits + (size_t)ROWS * VOCAB;

  const long long DD = (long long)D_MODEL * D_MODEL;

  // ---- pack all weights to fragment-major f16 (once per call) ----
  pack_b_kernel<<<dim3(1024, 1, NLAYER), 256, 0, stream>>>(
      Wq, pWq, D_MODEL, D_MODEL, HDIM, HBLK, WQ_LS, DD);
  pack_b_kernel<<<dim3(1024, 1, NLAYER), 256, 0, stream>>>(
      Wk, pWk, D_MODEL, D_MODEL, HDIM, HBLK, WQ_LS, DD);
  pack_b_kernel<<<dim3(1024, 1, NLAYER), 256, 0, stream>>>(
      Wv, pWv, D_MODEL, D_MODEL, HDIM, HBLK, WQ_LS, DD);
  pack_b_kernel<<<dim3(1024, 1, NLAYER), 256, 0, stream>>>(
      Wo, pWo, D_MODEL, D_MODEL, D_MODEL, 64, DD, DD);
  pack_b_kernel<<<dim3(2048, 1, NLAYER), 256, 0, stream>>>(
      W1, pW1, D_MODEL, 4 * D_MODEL, 4 * D_MODEL, 64, 4 * DD, 4 * DD);
  pack_b_kernel<<<dim3(2048, 1, NLAYER), 256, 0, stream>>>(
      W2, pW2, 4 * D_MODEL, D_MODEL, D_MODEL, 64, 4 * DD, 4 * DD);
  pack_b_kernel<<<dim3(4096, 1, 1), 256, 0, stream>>>(
      Wlm, pWlm, D_MODEL, VOCAB, VOCAB, 64, 0, 0);

  embed_kernel<<<ROWS, 256, 0, stream>>>(idx, tok_emb, pos_emb, x);

  for (int l = 0; l < NLAYER; ++l) {
    ln_kernel<<<ROWS, 256, 0, stream>>>(x, ln1_g + l * D_MODEL, ln1_b + l * D_MODEL, hbuf);

    // q/k/v = h @ W{q,k,v}[l]   (packed weights via TDM, f16 out)
    launch_gemm(stream, hbuf, pWq + (size_t)l * DD, nullptr, nullptr, q,
                ROWS, D_MODEL, D_MODEL, D_MODEL, 0, D_MODEL,
                0, 0, 0, 0, 0, 0, 1, 1, 128, 0, 0, 1, 0, 0, 0.f);
    launch_gemm(stream, hbuf, pWk + (size_t)l * DD, nullptr, nullptr, kbuf,
                ROWS, D_MODEL, D_MODEL, D_MODEL, 0, D_MODEL,
                0, 0, 0, 0, 0, 0, 1, 1, 128, 0, 0, 1, 0, 0, 0.f);
    launch_gemm(stream, hbuf, pWv + (size_t)l * DD, nullptr, nullptr, vbuf,
                ROWS, D_MODEL, D_MODEL, D_MODEL, 0, D_MODEL,
                0, 0, 0, 0, 0, 0, 1, 1, 128, 0, 0, 1, 0, 0, 0.f);

    // scores[b,h] = causal(Q_bh x K_bh^T * scale), f32 out  (z = b*H + h)
    launch_gemm(stream, q, kbuf, nullptr, nullptr, scores,
                SEQ, SEQ, HDIM, D_MODEL, D_MODEL, SEQ,
                (long long)SEQ * D_MODEL, HDIM,
                (long long)SEQ * D_MODEL, HDIM,
                (long long)NHEAD * SEQ * SEQ, (long long)SEQ * SEQ,
                NHEAD, BATCH * NHEAD, 128, 0, /*b_mode*/1, 0, 0, /*causal*/1, scale);

    softmax_kernel<<<BATCH * NHEAD * SEQ, 256, 0, stream>>>(scores, probs);

    // att[b,h] = P x V_bh   (f16 probs x f16 v, scalar-staged B, f16 out)
    launch_gemm(stream, probs, vbuf, nullptr, nullptr, att,
                SEQ, HDIM, SEQ, SEQ, D_MODEL, D_MODEL,
                (long long)NHEAD * SEQ * SEQ, (long long)SEQ * SEQ,
                (long long)SEQ * D_MODEL, HDIM,
                (long long)SEQ * D_MODEL, HDIM,
                NHEAD, BATCH * NHEAD, 64, 0, /*b_mode*/2, 1, 0, 0, 0.f);

    // x = x + att @ Wo + bo   (f32 residual out)
    launch_gemm(stream, att, pWo + (size_t)l * DD, bo + (size_t)l * D_MODEL, x, x,
                ROWS, D_MODEL, D_MODEL, D_MODEL, 0, D_MODEL,
                0, 0, 0, 0, 0, 0, 1, 1, 128, 0, 0, 0, 0, 0, 0.f);

    // MLP
    ln_kernel<<<ROWS, 256, 0, stream>>>(x, ln2_g + l * D_MODEL, ln2_b + l * D_MODEL, hbuf);
    launch_gemm(stream, hbuf, pW1 + (size_t)l * 4 * DD, b1 + (size_t)l * 4 * D_MODEL,
                nullptr, ffn,
                ROWS, 4 * D_MODEL, D_MODEL, D_MODEL, 0, 4 * D_MODEL,
                0, 0, 0, 0, 0, 0, 1, 1, 128, 0, 0, 1, /*relu*/1, 0, 0.f);
    launch_gemm(stream, ffn, pW2 + (size_t)l * 4 * DD, b2 + (size_t)l * D_MODEL, x, x,
                ROWS, D_MODEL, 4 * D_MODEL, 4 * D_MODEL, 0, D_MODEL,
                0, 0, 0, 0, 0, 0, 1, 1, 128, 0, 0, 0, 0, 0, 0.f);
  }

  // final LN + LM head -> logits (f32, straight into d_out)
  ln_kernel<<<ROWS, 256, 0, stream>>>(x, lnf_g, lnf_b, hbuf);
  launch_gemm(stream, hbuf, pWlm, blm, nullptr, logits,
              ROWS, VOCAB, D_MODEL, D_MODEL, 0, VOCAB,
              0, 0, 0, 0, 0, 0, 1, 1, 128, 0, 0, 0, 0, 0, 0.f);

  // cross-entropy loss
  zero_loss_kernel<<<1, 1, 0, stream>>>(lossp);
  loss_kernel<<<ROWS, 256, 0, stream>>>(logits, targets, lossp);
}
